// TraditionalMoE_40441412059787
// MI455X (gfx1250) — compile-verified, matrix operand order checked by reading, not run on hardware
//
#include <hip/hip_runtime.h>
#include <hip/hip_bf16.h>

// ---------------- problem constants ----------------
#define B_TOK   16384
#define S_DIM   128
#define H_DIM   1024
#define E_NUM   8
#define IN_DIM  131
#define INP_PAD 160      // pad K of layer-1 to multiple of 32
#define GH_DIM  32

typedef __attribute__((ext_vector_type(16))) __bf16 v16bf;
typedef __attribute__((ext_vector_type(8)))  float  v8f;
typedef int v4i __attribute__((vector_size(16)));   // matches builtin param pointee

union Frag { uint4 q[2]; v16bf v; };

static __device__ __forceinline__ unsigned short f2bf(float f) {
    __bf16 h = (__bf16)f;
    unsigned short u;
    __builtin_memcpy(&u, &h, 2);
    return u;
}

// ---------- gfx1250 async global->LDS path (guarded; falls back to sync copy) ----------
#if defined(__has_builtin)
#if __has_builtin(__builtin_amdgcn_global_load_async_to_lds_b128)
#define ASYNC_LDS 1
#endif
#endif
#ifndef ASYNC_LDS
#define ASYNC_LDS 0
#endif

#if ASYNC_LDS && defined(__has_builtin)
#if __has_builtin(__builtin_amdgcn_s_wait_asynccnt)
#define WAITA(n) __builtin_amdgcn_s_wait_asynccnt(n)
#else
#define WAITA(n) asm volatile("s_wait_asynccnt %0" :: "i"(n) : "memory")
#endif
#else
#define WAITA(n) ((void)0)
#endif

static __device__ __forceinline__ void cp16(const unsigned short* __restrict__ g,
                                            unsigned short* l)
{
#if ASYNC_LDS
    __builtin_amdgcn_global_load_async_to_lds_b128(
        (__attribute__((address_space(1))) v4i*)g,
        (__attribute__((address_space(3))) v4i*)l,
        0, 0);
#else
    *reinterpret_cast<uint4*>(l) = *reinterpret_cast<const uint4*>(g);
#endif
}

// ---------------- kernel 1: build bf16 input + gate MLP + softmax ----------------
__global__ void prep_gates_kernel(const float* __restrict__ t, const float* __restrict__ x,
                                  const float* __restrict__ G1, const float* __restrict__ g1b,
                                  const float* __restrict__ G2, const float* __restrict__ g2b,
                                  unsigned short* __restrict__ inp_bf, float* __restrict__ gates)
{
    int b = blockIdx.x * blockDim.x + threadIdx.x;
    if (b >= B_TOK) return;

    float tv = t[b];
    float sv = __sinf(tv);
    float cv = __cosf(tv);

    float h[GH_DIM];
#pragma unroll
    for (int j = 0; j < GH_DIM; ++j) h[j] = g1b[j];

    unsigned short* row = inp_bf + (size_t)b * INP_PAD;
    for (int i = 0; i < INP_PAD; ++i) {
        float v;
        if (i < S_DIM)       v = x[(size_t)b * S_DIM + i];
        else if (i == 128)   v = tv;
        else if (i == 129)   v = sv;
        else if (i == 130)   v = cv;
        else                 v = 0.0f;
        row[i] = f2bf(v);
        if (i < IN_DIM) {
#pragma unroll
            for (int j = 0; j < GH_DIM; ++j) h[j] = fmaf(v, G1[i * GH_DIM + j], h[j]);
        }
    }

    float logit[E_NUM];
#pragma unroll
    for (int e = 0; e < E_NUM; ++e) logit[e] = g2b[e];
#pragma unroll
    for (int j = 0; j < GH_DIM; ++j) {
        float hv = fmaxf(h[j], 0.0f);
#pragma unroll
        for (int e = 0; e < E_NUM; ++e) logit[e] = fmaf(hv, G2[j * E_NUM + e], logit[e]);
    }
    float m = logit[0];
#pragma unroll
    for (int e = 1; e < E_NUM; ++e) m = fmaxf(m, logit[e]);
    float s = 0.0f;
#pragma unroll
    for (int e = 0; e < E_NUM; ++e) { logit[e] = __expf(logit[e] - m); s += logit[e]; }
    float inv = 1.0f / s;
#pragma unroll
    for (int e = 0; e < E_NUM; ++e) gates[(size_t)b * E_NUM + e] = logit[e] * inv;
}

// ---------------- transpose + convert: W (E,K,N) f32 -> Wt (E,N,Kpad) bf16 ----------------
// 32x32 LDS-tiled: coalesced reads and writes. Rows k >= K are zero-filled (padding).
__global__ __launch_bounds__(256)
void cvt_t_kernel(const float* __restrict__ src, unsigned short* __restrict__ dst,
                  int K, int N, int Kpad)
{
    __shared__ float tile[32][33];
    const int e  = blockIdx.z;
    const int k0 = blockIdx.x * 32;     // Kpad tile origin
    const int n0 = blockIdx.y * 32;
    const int tx = threadIdx.x;         // 0..31
    const int ty = threadIdx.y;         // 0..7

    const float* s = src + (size_t)e * K * N;
#pragma unroll
    for (int j = 0; j < 32; j += 8) {
        int k = k0 + ty + j;
        tile[ty + j][tx] = (k < K) ? s[(size_t)k * N + n0 + tx] : 0.0f;
    }
    __syncthreads();

    unsigned short* d = dst + (size_t)e * N * Kpad;
#pragma unroll
    for (int j = 0; j < 32; j += 8) {
        int n = n0 + ty + j;
        d[(size_t)n * Kpad + k0 + tx] = f2bf(tile[tx][ty + j]);
    }
}

// ---------------- WMMA GEMM: C = act(A @ W + bias) ----------------
// A : (M=B_TOK, K) bf16 row-major (per expert via aStrideE; 0 = shared A)
// Wt: (N, K)     bf16 row-major per expert (pre-transposed weights)
// MODE 0: out = relu(..) -> bf16   MODE 1: out = (..) -> f32
// Block: 256 threads = 8 waves (4x2); tile 128x128, BK=32, double-buffered LDS,
// async global->LDS staging (ASYNCcnt). 4 async ops issued per wave per tile.
template<int MODE>
__global__ __launch_bounds__(256)
void gemm_kernel(const unsigned short* __restrict__ A, size_t aStrideE, int lda,
                 const unsigned short* __restrict__ Wt, size_t wStrideE,
                 const float* __restrict__ bias, int biasStrideE,
                 unsigned short* __restrict__ outB, float* __restrict__ outF,
                 size_t oStrideE, int K, int N)
{
    // [buf][A/B][128 rows x 32 bf16, padded to 40 ushorts (80 B = 5*16 B, bank-staggered)]
    constexpr int TILE_USH = 128 * 40;
    __shared__ __align__(16) unsigned short smem[2][2][TILE_USH];

    const int tid   = threadIdx.x;
    const int wave  = tid >> 5;
    const int lane  = tid & 31;
    const int waveR = wave >> 1;     // 0..3 : 32 rows each
    const int waveC = wave & 1;      // 0..1 : 64 cols each
    const int half  = lane >> 4;     // half-wave K sub-block select
    const int l16   = lane & 15;

    const int e = blockIdx.z;
    const unsigned short* Ae = A  + aStrideE * (size_t)e + (size_t)(blockIdx.y * 128) * lda;
    const unsigned short* Be = Wt + wStrideE * (size_t)e + (size_t)(blockIdx.x * 128) * K;

    v8f acc[2][4];
#pragma unroll
    for (int mt = 0; mt < 2; ++mt)
#pragma unroll
        for (int nt = 0; nt < 4; ++nt)
#pragma unroll
            for (int r = 0; r < 8; ++r) acc[mt][nt][r] = 0.0f;

    // stage one 128x32 A-tile and 128x32 B-tile: 2 x 512 chunks of 16 B;
    // per wave: 2 loop iters x 2 cp16 = 4 async instructions per tile.
    auto issue_tile = [&](int kt, int buf) {
        const int k0 = kt * 32;
#pragma unroll
        for (int c = tid; c < 512; c += 256) {
            const int row = c >> 2;
            const int kc  = c & 3;
            cp16(Ae + (size_t)row * lda + k0 + kc * 8, &smem[buf][0][row * 40 + kc * 8]);
            cp16(Be + (size_t)row * K   + k0 + kc * 8, &smem[buf][1][row * 40 + kc * 8]);
        }
    };

    const int nk = K >> 5;
    issue_tile(0, 0);

    for (int kt = 0; kt < nk; ++kt) {
        const int cur = kt & 1;
        if (kt + 1 < nk) {
            issue_tile(kt + 1, cur ^ 1);   // prefetch next tile (+4 async ops/wave)
            WAITA(4);                      // in-order: <=4 outstanding => current tile landed
        } else {
            WAITA(0);
        }
        __syncthreads();

        const unsigned short* sA = smem[cur][0];
        const unsigned short* sB = smem[cur][1];

        Frag af[2], bf[4];
#pragma unroll
        for (int mt = 0; mt < 2; ++mt) {
            int rowA = waveR * 32 + mt * 16 + l16;
            af[mt].q[0] = *reinterpret_cast<const uint4*>(&sA[rowA * 40 + half * 8]);
            af[mt].q[1] = *reinterpret_cast<const uint4*>(&sA[rowA * 40 + 16 + half * 8]);
        }
#pragma unroll
        for (int nt = 0; nt < 4; ++nt) {
            int colB = waveC * 64 + nt * 16 + l16;
            bf[nt].q[0] = *reinterpret_cast<const uint4*>(&sB[colB * 40 + half * 8]);
            bf[nt].q[1] = *reinterpret_cast<const uint4*>(&sB[colB * 40 + 16 + half * 8]);
        }

#pragma unroll
        for (int mt = 0; mt < 2; ++mt)
#pragma unroll
            for (int nt = 0; nt < 4; ++nt)
                acc[mt][nt] = __builtin_amdgcn_wmma_f32_16x16x32_bf16(
                    false, af[mt].v, false, bf[nt].v, (short)0, acc[mt][nt], false, false);

        __syncthreads();   // protect buf `cur` before next iteration's prefetch overwrites it
    }

    // ---- epilogue: bias (+relu) ----
    const float* be = bias + (size_t)biasStrideE * e;
    const size_t obase = oStrideE * (size_t)e;
#pragma unroll
    for (int mt = 0; mt < 2; ++mt) {
#pragma unroll
        for (int nt = 0; nt < 4; ++nt) {
            int colN = blockIdx.x * 128 + waveC * 64 + nt * 16 + l16;
            float bv = be[colN];
#pragma unroll
            for (int r = 0; r < 8; ++r) {
                int rowG = blockIdx.y * 128 + waveR * 32 + mt * 16 + half * 8 + r;
                float v = acc[mt][nt][r] + bv;
                if (MODE == 0) {
                    v = fmaxf(v, 0.0f);
                    outB[obase + (size_t)rowG * N + colN] = f2bf(v);
                } else {
                    outF[obase + (size_t)rowG * N + colN] = v;
                }
            }
        }
    }
}

// ---------------- finalize: per-token norm clip + gated expert sum ----------------
// one wave (32 lanes) per token; 4 f32 of S per lane
__global__ __launch_bounds__(256)
void finalize_kernel(const float* __restrict__ out_e, const float* __restrict__ gates,
                     float* __restrict__ out)
{
    int wid  = (blockIdx.x * blockDim.x + threadIdx.x) >> 5;   // token
    int lane = threadIdx.x & 31;
    if (wid >= B_TOK) return;

    float4 accv = make_float4(0.f, 0.f, 0.f, 0.f);
#pragma unroll
    for (int e = 0; e < E_NUM; ++e) {
        float4 v = *reinterpret_cast<const float4*>(
            out_e + ((size_t)e * B_TOK + wid) * S_DIM + lane * 4);
        float ss = v.x * v.x + v.y * v.y + v.z * v.z + v.w * v.w;
#pragma unroll
        for (int off = 16; off > 0; off >>= 1) ss += __shfl_xor(ss, off, 32);
        float norm = sqrtf(ss);
        float sc   = fminf(1.0f, 10.0f / (norm + 1e-6f));
        float g    = gates[(size_t)wid * E_NUM + e] * sc;
        accv.x = fmaf(v.x, g, accv.x);
        accv.y = fmaf(v.y, g, accv.y);
        accv.z = fmaf(v.z, g, accv.z);
        accv.w = fmaf(v.w, g, accv.w);
    }
    *reinterpret_cast<float4*>(out + (size_t)wid * S_DIM + lane * 4) = accv;
}

// ---------------- launcher ----------------
extern "C" void kernel_launch(void* const* d_in, const int* in_sizes, int n_in,
                              void* d_out, int out_size, void* d_ws, size_t ws_size,
                              hipStream_t stream)
{
    (void)in_sizes; (void)n_in; (void)out_size; (void)ws_size;

    const float* t   = (const float*)d_in[0];
    const float* x   = (const float*)d_in[1];
    const float* W1  = (const float*)d_in[2];
    const float* b1  = (const float*)d_in[3];
    const float* W2  = (const float*)d_in[4];
    const float* b2  = (const float*)d_in[5];
    const float* W3  = (const float*)d_in[6];
    const float* b3  = (const float*)d_in[7];
    const float* G1  = (const float*)d_in[8];
    const float* g1b = (const float*)d_in[9];
    const float* G2  = (const float*)d_in[10];
    const float* g2b = (const float*)d_in[11];
    float* out = (float*)d_out;

    char* ws = (char*)d_ws;
    size_t off = 0;
    auto carve = [&](size_t bytes) -> void* {
        void* p = ws + off;
        off += (bytes + 255) & ~(size_t)255;
        return p;
    };

    unsigned short* inp_bf = (unsigned short*)carve((size_t)B_TOK * INP_PAD * 2);
    float*          gates  = (float*)         carve((size_t)B_TOK * E_NUM * 4);
    unsigned short* w1t    = (unsigned short*)carve((size_t)E_NUM * H_DIM * INP_PAD * 2); // (E,N=H,K=160)
    unsigned short* w2t    = (unsigned short*)carve((size_t)E_NUM * H_DIM * H_DIM * 2);   // (E,N=H,K=H)
    unsigned short* w3t    = (unsigned short*)carve((size_t)E_NUM * S_DIM * H_DIM * 2);   // (E,N=S,K=H)
    unsigned short* h1     = (unsigned short*)carve((size_t)E_NUM * B_TOK * H_DIM * 2);
    unsigned short* h2     = (unsigned short*)carve((size_t)E_NUM * B_TOK * H_DIM * 2);
    float*          out_e  = (float*)h1;   // alias: h1 is dead once GEMM2 completes

    // 1) input assembly + gates
    prep_gates_kernel<<<B_TOK / 256, 256, 0, stream>>>(t, x, G1, g1b, G2, g2b, inp_bf, gates);

    // 2) weight convert + transpose: W (K,N) f32 -> Wt (N,Kpad) bf16
    dim3 tb(32, 8);
    cvt_t_kernel<<<dim3(INP_PAD / 32, H_DIM / 32, E_NUM), tb, 0, stream>>>(W1, w1t, IN_DIM, H_DIM, INP_PAD);
    cvt_t_kernel<<<dim3(H_DIM / 32, H_DIM / 32, E_NUM), tb, 0, stream>>>(W2, w2t, H_DIM, H_DIM, H_DIM);
    cvt_t_kernel<<<dim3(H_DIM / 32, S_DIM / 32, E_NUM), tb, 0, stream>>>(W3, w3t, H_DIM, S_DIM, H_DIM);

    // 3) layer 1: h1 = relu(inp @ W1 + b1)   K=160, N=1024, A shared across experts
    gemm_kernel<0><<<dim3(H_DIM / 128, B_TOK / 128, E_NUM), 256, 0, stream>>>(
        inp_bf, (size_t)0, INP_PAD,
        w1t, (size_t)H_DIM * INP_PAD,
        b1, H_DIM,
        h1, nullptr, (size_t)B_TOK * H_DIM,
        INP_PAD, H_DIM);

    // 4) layer 2: h2 = relu(h1 @ W2 + b2)    K=1024, N=1024 (dominant GEMM)
    gemm_kernel<0><<<dim3(H_DIM / 128, B_TOK / 128, E_NUM), 256, 0, stream>>>(
        h1, (size_t)B_TOK * H_DIM, H_DIM,
        w2t, (size_t)H_DIM * H_DIM,
        b2, H_DIM,
        h2, nullptr, (size_t)B_TOK * H_DIM,
        H_DIM, H_DIM);

    // 5) layer 3: out_e = h2 @ W3 + b3       K=1024, N=128, f32 out
    gemm_kernel<1><<<dim3(S_DIM / 128, B_TOK / 128, E_NUM), 256, 0, stream>>>(
        h2, (size_t)B_TOK * H_DIM, H_DIM,
        w3t, (size_t)S_DIM * H_DIM,
        b3, S_DIM,
        nullptr, out_e, (size_t)B_TOK * S_DIM,
        H_DIM, S_DIM);

    // 6) norm clip + gated sum over experts
    finalize_kernel<<<B_TOK / 8, 256, 0, stream>>>(out_e, gates, out);
}
